// EmbeddingM_44590350467213
// MI455X (gfx1250) — matrix operand order repaired and checked by reading, not compiled, not run on hardware
//
#include <hip/hip_runtime.h>
#include <hip/hip_bf16.h>

typedef __attribute__((ext_vector_type(16))) _Float16 v16h;
typedef __attribute__((ext_vector_type(8)))  float    v8f;

#define FMDIM 128
#define LDSPAD 8
#define MTILE 64

static inline int cdiv(int a, int b) { return (a + b - 1) / b; }

// ---------------------------------------------------------------------------
// Small elementwise helpers
// ---------------------------------------------------------------------------
__global__ void zero6_kernel(float* p) {
    if (threadIdx.x < 6) p[threadIdx.x] = 0.0f;
}

__global__ void fill1_kernel(float* p, int n) {
    int i = blockIdx.x * blockDim.x + threadIdx.x;
    if (i < n) p[i] = 1.0f;   // self-loop weight contributes 1 to every degree
}

__global__ void deg_accum_kernel(const int* __restrict__ edges,
                                 const float* __restrict__ ew,
                                 float* __restrict__ deg, int E) {
    int e = blockIdx.x * blockDim.x + threadIdx.x;
    if (e < E) atomicAdd(&deg[edges[E + e]], ew[e]);
}

__global__ void rsqrt_kernel(float* __restrict__ d, int n) {
    int i = blockIdx.x * blockDim.x + threadIdx.x;
    if (i < n) {
        float v = d[i];
        d[i] = (v > 0.0f) ? rsqrtf(fmaxf(v, 1e-12f)) : 0.0f;
    }
}

// ---------------------------------------------------------------------------
// Edge aggregation: one wave32 per edge, float4 per lane (32*4 = 128 feats).
// Per-edge metadata is wave-uniform -> force into SGPRs via readfirstlane.
// ---------------------------------------------------------------------------
__global__ __launch_bounds__(256) void agg_kernel(const int* __restrict__ edges,
                                                  const float* __restrict__ ew,
                                                  const float* __restrict__ dis,
                                                  const float* __restrict__ h,
                                                  float* __restrict__ L, int E) {
    int wid  = (blockIdx.x * blockDim.x + threadIdx.x) >> 5;
    int lane = threadIdx.x & 31;
    if (wid >= E) return;
    int   s  = __builtin_amdgcn_readfirstlane(edges[wid]);
    int   d  = __builtin_amdgcn_readfirstlane(edges[E + wid]);
    float w  = __builtin_amdgcn_readfirstlane(__float_as_uint(ew[wid]));
    float nrm = dis[s] * __uint_as_float((unsigned)__float_as_uint(w)) * dis[d];
    float4 v  = reinterpret_cast<const float4*>(h + (size_t)s * FMDIM)[lane];
    float* o  = L + (size_t)d * FMDIM + lane * 4;
    atomicAdd(o + 0, nrm * v.x);
    atomicAdd(o + 1, nrm * v.y);
    atomicAdd(o + 2, nrm * v.z);
    atomicAdd(o + 3, nrm * v.w);
}

// ---------------------------------------------------------------------------
// Fused bias + ReLU (in place) + global-avg-pool partial sum
// ---------------------------------------------------------------------------
__global__ __launch_bounds__(256) void bias_relu_pool_kernel(float* __restrict__ L,
                                                             const float* __restrict__ b,
                                                             float* __restrict__ pooled_slot,
                                                             int total) {
    int idx = blockIdx.x * blockDim.x + threadIdx.x;
    float v = 0.0f;
    if (idx < total) {
        float val = L[idx] + b[idx & (FMDIM - 1)];
        val = fmaxf(val, 0.0f);
        L[idx] = val;
        v = val;
    }
    __shared__ float red[256];
    red[threadIdx.x] = v;
    __syncthreads();
    for (int s = 128; s > 0; s >>= 1) {
        if ((int)threadIdx.x < s) red[threadIdx.x] += red[threadIdx.x + s];
        __syncthreads();
    }
    if (threadIdx.x == 0) atomicAdd(pooled_slot, red[0]);
}

// ---------------------------------------------------------------------------
// WMMA GEMM: H[M,128] = A[M,128] @ W[128,128], fp32 in/out, f16 matrix cores.
// Fused self-loop init:  L[m,f] = dis[m]^2 * H[m,f]  written alongside H.
// One block = 256 threads = 8 waves; MTILE=64 rows; wave w owns N-cols w*16..
// W staged TRANSPOSED in LDS so B-fragment reads are contiguous (ds_load_b128).
// ---------------------------------------------------------------------------
__global__ __launch_bounds__(256) void gemm_wmma_kernel(const float* __restrict__ A,
                                                        const float* __restrict__ W,
                                                        const float* __restrict__ dis,
                                                        float* __restrict__ H,
                                                        float* __restrict__ L, int M) {
    __shared__ _Float16 WsT[FMDIM][FMDIM + LDSPAD];  // WsT[n][k] = W[k][n]
    __shared__ _Float16 As[MTILE][FMDIM + LDSPAD];
    const int tid = threadIdx.x;
    const int m0  = blockIdx.x * MTILE;

    // Stage W tile transposed (128x128 f32 -> f16): 4096 float4 reads.
    for (int i = tid; i < FMDIM * (FMDIM / 4); i += 256) {
        int r = i >> 5, c4 = (i & 31) * 4;        // W row r, cols c4..c4+3
        float4 wv = reinterpret_cast<const float4*>(W)[i];
        WsT[c4 + 0][r] = (_Float16)wv.x;  WsT[c4 + 1][r] = (_Float16)wv.y;
        WsT[c4 + 2][r] = (_Float16)wv.z;  WsT[c4 + 3][r] = (_Float16)wv.w;
    }
    // Stage A tile (64x128 f32 -> f16), zero-pad out-of-range rows.
    for (int i = tid; i < MTILE * (FMDIM / 4); i += 256) {
        int r = i >> 5, c4 = (i & 31) * 4;
        int row = m0 + r;
        float4 av = make_float4(0.f, 0.f, 0.f, 0.f);
        if (row < M) av = reinterpret_cast<const float4*>(A)[(size_t)row * 32 + (i & 31)];
        As[r][c4 + 0] = (_Float16)av.x;  As[r][c4 + 1] = (_Float16)av.y;
        As[r][c4 + 2] = (_Float16)av.z;  As[r][c4 + 3] = (_Float16)av.w;
    }
    __syncthreads();

    const int wave = tid >> 5;
    const int lane = tid & 31;
    const int nb   = wave * 16;          // this wave's 16 output columns
    const int lhi  = (lane >= 16) ? 1 : 0;
    const int lm   = lane & 15;
    const bool full = (m0 + MTILE) <= M;

    // B fragments (32x16 f16): lane = N col, element e -> K = kbase + 16*lhi + e
    // -> 32 contiguous bytes per lane in WsT => vectorized ds_load_b128.
    v16h bfrag[4];
#pragma unroll
    for (int kk = 0; kk < 4; ++kk) {
#pragma unroll
        for (int e = 0; e < 16; ++e)
            bfrag[kk][e] = WsT[nb + lm][kk * 32 + lhi * 16 + e];
    }

#pragma unroll
    for (int sub = 0; sub < MTILE / 16; ++sub) {
        const int mb = sub * 16;
        v8f c = {};
#pragma unroll
        for (int kk = 0; kk < 4; ++kk) {
            // A fragment (16x32 f16): one contiguous 32B run per lane (K = 16*lhi + e).
            v16h a;
#pragma unroll
            for (int e = 0; e < 16; ++e) {
                int kloc = e + lhi * 16;
                a[e] = As[mb + lm][kk * 32 + kloc];
            }
            c = __builtin_amdgcn_wmma_f32_16x16x32_f16(
                    false, a, false, bfrag[kk], (short)0, c, false, false);
        }
        // D layout: VGPR r, lanes 0-15 -> M = mb+r, lanes 16-31 -> M = mb+r+8
        if (full) {
#pragma unroll
            for (int r = 0; r < 8; ++r) {
                int mrow = m0 + mb + r + lhi * 8;
                size_t off = (size_t)mrow * FMDIM + nb + lm;
                float ds = dis[mrow];
                H[off] = c[r];
                L[off] = ds * ds * c[r];
            }
        } else {
#pragma unroll
            for (int r = 0; r < 8; ++r) {
                int mrow = m0 + mb + r + lhi * 8;
                if (mrow < M) {
                    size_t off = (size_t)mrow * FMDIM + nb + lm;
                    float ds = dis[mrow];
                    H[off] = c[r];
                    L[off] = ds * ds * c[r];
                }
            }
        }
    }
}

// ---------------------------------------------------------------------------
// SE attention head: a = sigmoid(relu(pooled @ fc1_w.T + fc1_b) @ fc2_w.T + fc2_b)
// ---------------------------------------------------------------------------
__global__ void se_fc_kernel(const float* __restrict__ pooled,
                             const float* __restrict__ fc1_w, const float* __restrict__ fc1_b,
                             const float* __restrict__ fc2_w, const float* __restrict__ fc2_b,
                             float* __restrict__ a_out, float inv_count) {
    __shared__ float pm[6];
    __shared__ float hid[30];
    int t = threadIdx.x;
    if (t < 6) pm[t] = pooled[t] * inv_count;
    __syncthreads();
    if (t < 30) {
        float s = fc1_b[t];
#pragma unroll
        for (int c = 0; c < 6; ++c) s += pm[c] * fc1_w[t * 6 + c];
        hid[t] = fmaxf(s, 0.0f);
    }
    __syncthreads();
    if (t < 6) {
        float s = fc2_b[t];
        for (int j = 0; j < 30; ++j) s += hid[j] * fc2_w[t * 30 + j];
        a_out[t] = 1.0f / (1.0f + __expf(-s));
    }
}

// out[n,f] = sum_c cnn_w[c] * relu(a[c] * L_c[n,f]) + cnn_b[0]
__global__ __launch_bounds__(256) void combine_kernel(const float* __restrict__ Lbase, int NF,
                                                      const float* __restrict__ a,
                                                      const float* __restrict__ cnn_w,
                                                      const float* __restrict__ cnn_b,
                                                      float* __restrict__ out, int total) {
    int idx = blockIdx.x * blockDim.x + threadIdx.x;
    if (idx >= total) return;
    float acc = cnn_b[0];
#pragma unroll
    for (int c = 0; c < 6; ++c)
        acc += cnn_w[c] * fmaxf(a[c] * Lbase[(size_t)c * NF + idx], 0.0f);
    out[idx] = acc;
}

// ---------------------------------------------------------------------------
extern "C" void kernel_launch(void* const* d_in, const int* in_sizes, int n_in,
                              void* d_out, int out_size, void* d_ws, size_t ws_size,
                              hipStream_t stream) {
    (void)n_in; (void)out_size; (void)ws_size;

    const float* x_m = (const float*)d_in[0];
    const int*   edges[3] = { (const int*)d_in[1], (const int*)d_in[3], (const int*)d_in[5] };
    const float* ew[3]    = { (const float*)d_in[2], (const float*)d_in[4], (const float*)d_in[6] };
    const float* Wm[6]; const float* bm[6];
    for (int i = 0; i < 6; ++i) {
        Wm[i] = (const float*)d_in[7 + 2 * i];
        bm[i] = (const float*)d_in[8 + 2 * i];
    }
    const float* fc1_w = (const float*)d_in[19];
    const float* fc1_b = (const float*)d_in[20];
    const float* fc2_w = (const float*)d_in[21];
    const float* fc2_b = (const float*)d_in[22];
    const float* cnn_w = (const float*)d_in[23];
    const float* cnn_b = (const float*)d_in[24];
    float* out = (float*)d_out;

    const int N  = in_sizes[0] / FMDIM;   // 50000 nodes
    const int E  = in_sizes[2];           // 800000 edges per view
    const int NF = N * FMDIM;             // 6.4M elements per map

    // Workspace layout (floats): 6 layer maps | h | dis[3*N] | pooled[6] | a[6]
    float* ws     = (float*)d_ws;
    float* Lbase  = ws;
    float* h      = ws + (size_t)6 * NF;
    float* disbuf = ws + (size_t)7 * NF;
    float* pooled = disbuf + (size_t)3 * N;
    float* attn   = pooled + 6;

    zero6_kernel<<<1, 32, 0, stream>>>(pooled);

    for (int v = 0; v < 3; ++v) {
        float* dis = disbuf + (size_t)v * N;
        // degree (with self loop) -> dis = deg^-1/2
        fill1_kernel<<<cdiv(N, 256), 256, 0, stream>>>(dis, N);
        deg_accum_kernel<<<cdiv(E, 256), 256, 0, stream>>>(edges[v], ew[v], dis, E);
        rsqrt_kernel<<<cdiv(N, 256), 256, 0, stream>>>(dis, N);

        const float* in = x_m;
        for (int layer = 0; layer < 2; ++layer) {
            int    c = 2 * v + layer;
            float* L = Lbase + (size_t)c * NF;
            // GEMM h = in @ W, fused with L = dis^2 * h (self-loop init)
            gemm_wmma_kernel<<<cdiv(N, MTILE), 256, 0, stream>>>(in, Wm[c], dis, h, L, N);
            agg_kernel<<<cdiv(E * 32, 256), 256, 0, stream>>>(edges[v], ew[v], dis, h, L, E);
            bias_relu_pool_kernel<<<cdiv(NF, 256), 256, 0, stream>>>(L, bm[c], pooled + c, NF);
            in = L;
        }
    }

    se_fc_kernel<<<1, 32, 0, stream>>>(pooled, fc1_w, fc1_b, fc2_w, fc2_b,
                                       attn, 1.0f / (float)NF);
    combine_kernel<<<cdiv(NF, 256), 256, 0, stream>>>(Lbase, NF, attn, cnn_w, cnn_b, out, NF);
}
